// Attn_77111842832758
// MI455X (gfx1250) — matrix operand order repaired and checked by reading, not compiled, not run on hardware
//
#include <hip/hip_runtime.h>
#include <hip/hip_bf16.h>
#include <math.h>

typedef __attribute__((ext_vector_type(16))) __bf16 v16bf;
typedef __attribute__((ext_vector_type(8)))  float  v8f;
typedef int v4i_ __attribute__((vector_size(16)));
typedef __attribute__((address_space(1))) v4i_* gv4i_p;   // global int4*
typedef __attribute__((address_space(3))) v4i_* lv4i_p;   // LDS int4*

#define H       1024
#define BATCH   32
#define TLEN    1024
#define TWO_H   2048
#define WAVES   8
#define BROW_BYTES   2064                       // (1024+8) bf16 padded row: 516 dw = 4 mod 64 banks
#define BPANEL_BYTES (16 * BROW_BYTES)          // 33 024 B per 16-g panel
#define SMEM_BYTES   (2 * BPANEL_BYTES)         // 66 048 B double buffer -> up to 4 WGs / WGP

#if __has_builtin(__builtin_amdgcn_tanhf)
__device__ __forceinline__ float fast_tanhf(float x) { return __builtin_amdgcn_tanhf(x); }
#else
__device__ __forceinline__ float fast_tanhf(float x) { return tanhf(x); }
#endif

#if __has_builtin(__builtin_amdgcn_global_load_async_to_lds_b128) && \
    __has_builtin(__builtin_amdgcn_s_wait_asynccnt)
#define USE_ASYNC_LDS 1
#define ASYNC_WAIT()  __builtin_amdgcn_s_wait_asynccnt(0)
#else
#define USE_ASYNC_LDS 0
#define ASYNC_WAIT()
#endif

__device__ __forceinline__ unsigned int pack_bf16x2(float lo, float hi) {
  unsigned int a = __float_as_uint(lo);
  unsigned int b = __float_as_uint(hi);
  a = (a + 0x7FFFu + ((a >> 16) & 1u)) >> 16;   // RNE fp32 -> bf16
  b = (b + 0x7FFFu + ((b >> 16) & 1u)) >> 16;
  return a | (b << 16);
}

// ---- Kernel 0a: enc (fp32) -> encB (bf16, row-major) -------------------------------
__global__ void enc_convert(const float* __restrict__ enc, unsigned short* __restrict__ encB) {
  size_t idx = ((size_t)blockIdx.x * 256 + threadIdx.x) * 4;
  float4 f = *reinterpret_cast<const float4*>(enc + idx);
  uint2 o;
  o.x = pack_bf16x2(f.x, f.y);
  o.y = pack_bf16x2(f.z, f.w);
  *reinterpret_cast<uint2*>(encB + idx) = o;
}

// ---- Kernel 0b: W[:, H:2H] (fp32) -> WeB (bf16, row-major g x h) -------------------
__global__ void we_convert(const float* __restrict__ W, unsigned short* __restrict__ WeB) {
  int idx = blockIdx.x * 256 + threadIdx.x;      // one float4 per thread
  int g  = idx >> 8;
  int h4 = (idx & 255) * 4;
  float4 f = *reinterpret_cast<const float4*>(W + (size_t)g * TWO_H + H + h4);
  uint2 o;
  o.x = pack_bf16x2(f.x, f.y);
  o.y = pack_bf16x2(f.z, f.w);
  *reinterpret_cast<uint2*>(WeB + (size_t)g * H + h4) = o;
}

// ---- Kernel 1: s[b,g] = bias[g] + dot(hidden[b,:], W[g, 0:H]) ----------------------
__global__ void compute_s(const float* __restrict__ hidden, const float* __restrict__ W,
                          const float* __restrict__ bias, float* __restrict__ s_ws) {
  int wid  = (blockIdx.x * 256 + threadIdx.x) >> 5;   // one wave per output
  int lane = threadIdx.x & 31;
  int b = wid >> 10;
  int g = wid & 1023;
  const float4* wrow = reinterpret_cast<const float4*>(W + (size_t)g * TWO_H);
  const float4* hrow = reinterpret_cast<const float4*>(hidden + (size_t)b * H);
  float sum = 0.f;
#pragma unroll
  for (int i = 0; i < 8; ++i) {
    float4 w4 = wrow[lane + i * 32];
    float4 h4 = hrow[lane + i * 32];
    sum += w4.x * h4.x + w4.y * h4.y + w4.z * h4.z + w4.w * h4.w;
  }
#pragma unroll
  for (int ofs = 16; ofs > 0; ofs >>= 1) sum += __shfl_xor(sum, ofs, 32);
  if (lane == 0) s_ws[b * H + g] = sum + bias[g];
}

// ---- Kernel 2: fused bf16-WMMA GEMM + hw-tanh + v-dot -> energies[b,t] -------------
// 8 waves/block, one 16-row t-tile per wave; A read from global (WGP$-resident 32 KB
// tile), WeB panels double-buffered in LDS via async-to-LDS DMA.
__global__ __launch_bounds__(256) void attn_main(
    const unsigned short* __restrict__ encB, const unsigned short* __restrict__ WeB,
    const float* __restrict__ s_ws, const float* __restrict__ vvec,
    float* __restrict__ energies) {
  extern __shared__ char smem[];
  const int tid  = threadIdx.x;
  const int wave = tid >> 5;
  const int lane = tid & 31;
  const int m    = lane & 15;      // A: M row / B: N column
  const int half = lane >> 4;      // K-half selector per ISA fragment layout
  const int b    = blockIdx.x >> 3;
  const int t0   = (blockIdx.x & 7) * (WAVES * 16) + wave * 16;

  const char* Arow = reinterpret_cast<const char*>(
      encB + ((size_t)b * TLEN + t0 + m) * H);       // this lane's A row (2 KB)

  const int gr = tid >> 4;    // panel staging: row
  const int j  = tid & 15;    // panel staging: chunk

  // Stage one 16-row WeB panel (32 KB bf16) into LDS buffer `dst`.
  auto stage = [&](int gt, char* dst) {
    const unsigned short* srow = WeB + ((size_t)gt * 16 + gr) * H;
#pragma unroll
    for (int i = 0; i < 8; ++i) {
      int c = j * 8 + i * 128;
#if USE_ASYNC_LDS
      __builtin_amdgcn_global_load_async_to_lds_b128(
          (gv4i_p)(srow + c),
          (lv4i_p)(dst + gr * BROW_BYTES + c * 2),
          0, 0);
#else
      uint4 d = *reinterpret_cast<const uint4*>(srow + c);
      *reinterpret_cast<uint4*>(dst + gr * BROW_BYTES + c * 2) = d;
#endif
    }
  };

  float e[8];
#pragma unroll
  for (int r = 0; r < 8; ++r) e[r] = 0.f;

  stage(0, smem);
  ASYNC_WAIT();
  __syncthreads();

  for (int gt = 0; gt < 64; ++gt) {
    char* cur = smem + (gt & 1) * BPANEL_BYTES;
    if (gt + 1 < 64) stage(gt + 1, smem + ((gt + 1) & 1) * BPANEL_BYTES);  // overlap

    const float sv = s_ws[b * H + gt * 16 + m];   // hoisted: hide under wmma stream
    const float vv = vvec[gt * 16 + m];
    const char* Brow = cur + m * BROW_BYTES;

    v8f acc = {0.f, 0.f, 0.f, 0.f, 0.f, 0.f, 0.f, 0.f};
#pragma unroll 4
    for (int ks = 0; ks < 32; ++ks) {
      const int k0 = ks * 32;
      union { uint4 u[2]; v16bf f; } fa, fb;
      // A 16x32 bf16 fragment (global): lanes 0-15 K[k0,k0+8)+[k0+16,k0+24),
      //                                 lanes 16-31 shifted by 8.
      const int aoff = (k0 + half * 8) * 2;
      fa.u[0] = *reinterpret_cast<const uint4*>(Arow + aoff);
      fa.u[1] = *reinterpret_cast<const uint4*>(Arow + aoff + 32);
      // B 32x16 bf16 fragment (LDS): lane n holds K[k0+8*half,+8)+[k0+16+8*half,+8) of col n.
      const int boff = (k0 + half * 8) * 2;
      fb.u[0] = *reinterpret_cast<const uint4*>(Brow + boff);
      fb.u[1] = *reinterpret_cast<const uint4*>(Brow + boff + 32);
      acc = __builtin_amdgcn_wmma_f32_16x16x32_bf16(false, fa.f, false, fb.f,
                                                    (short)0, acc, false, false);
    }

    // Fused epilogue: e[r] += v[g] * tanh(s[b,g] + E); N = m, M = r + 8*half
#pragma unroll
    for (int r = 0; r < 8; ++r) e[r] += vv * fast_tanhf(sv + acc[r]);

    ASYNC_WAIT();        // next panel landed
    __syncthreads();     // all waves done with `cur`, next panel visible
  }

  // Reduce the N dimension across each 16-lane half-wave.
#pragma unroll
  for (int r = 0; r < 8; ++r) {
    float x = e[r];
    x += __shfl_xor(x, 1, 16);
    x += __shfl_xor(x, 2, 16);
    x += __shfl_xor(x, 4, 16);
    x += __shfl_xor(x, 8, 16);
    e[r] = x;
  }
  if (m == 0) {   // lane 0 -> M=0..7, lane 16 -> M=8..15
    float* dst = energies + (size_t)b * TLEN + t0 + half * 8;
#pragma unroll
    for (int r = 0; r < 8; ++r) dst[r] = e[r];
  }
}

// ---- Kernel 3: row softmax over T ---------------------------------------------------
__global__ void softmax_rows(const float* __restrict__ e, float* __restrict__ out) {
  __shared__ float red[8];
  __shared__ float bc;
  const int b = blockIdx.x;
  const int tid = threadIdx.x;
  const int wave = tid >> 5, lane = tid & 31;
  float4 x = reinterpret_cast<const float4*>(e + (size_t)b * TLEN)[tid];
  float mx = fmaxf(fmaxf(x.x, x.y), fmaxf(x.z, x.w));
  for (int ofs = 16; ofs > 0; ofs >>= 1) mx = fmaxf(mx, __shfl_xor(mx, ofs, 32));
  if (lane == 0) red[wave] = mx;
  __syncthreads();
  if (tid == 0) {
    float t = red[0];
    for (int i = 1; i < 8; ++i) t = fmaxf(t, red[i]);
    bc = t;
  }
  __syncthreads();
  const float M = bc;
  float4 ex;
  ex.x = __expf(x.x - M); ex.y = __expf(x.y - M);
  ex.z = __expf(x.z - M); ex.w = __expf(x.w - M);
  float sm = ex.x + ex.y + ex.z + ex.w;
  for (int ofs = 16; ofs > 0; ofs >>= 1) sm += __shfl_xor(sm, ofs, 32);
  if (lane == 0) red[wave] = sm;
  __syncthreads();
  if (tid == 0) {
    float t = 0.f;
    for (int i = 0; i < 8; ++i) t += red[i];
    bc = t;
  }
  __syncthreads();
  const float inv = 1.0f / bc;
  ex.x *= inv; ex.y *= inv; ex.z *= inv; ex.w *= inv;
  reinterpret_cast<float4*>(out + (size_t)b * TLEN)[tid] = ex;
}

extern "C" void kernel_launch(void* const* d_in, const int* in_sizes, int n_in,
                              void* d_out, int out_size, void* d_ws, size_t ws_size,
                              hipStream_t stream) {
  const float* hidden = (const float*)d_in[0];   // (1, 32, 1024)
  const float* enc    = (const float*)d_in[1];   // (32, 1024, 1024)
  const float* W      = (const float*)d_in[2];   // (1024, 2048)
  const float* bias   = (const float*)d_in[3];   // (1024,)
  const float* vvec   = (const float*)d_in[4];   // (1024,)
  float* out = (float*)d_out;                    // (32, 1, 1024)

  // Workspace (needs ~66.3 MB): encB 64 MB | WeB 2 MB | s 128 KB | energies 128 KB
  unsigned short* encB = (unsigned short*)d_ws;
  unsigned short* WeB  = (unsigned short*)((char*)d_ws + (size_t)64 * 1024 * 1024);
  float* s_ws     = (float*)((char*)d_ws + (size_t)66 * 1024 * 1024);
  float* energies = (float*)((char*)d_ws + (size_t)66 * 1024 * 1024 + 128 * 1024);

  enc_convert <<<32768, 256, 0, stream>>>(enc, encB);
  we_convert  <<<1024, 256, 0, stream>>>(W, WeB);
  compute_s   <<<4096, 256, 0, stream>>>(hidden, W, bias, s_ws);
  attn_main   <<<BATCH * (TLEN / (WAVES * 16)), 256, SMEM_BYTES, stream>>>(
                  encB, WeB, s_ws, vvec, energies);
  softmax_rows<<<BATCH, 256, 0, stream>>>(energies, out);
}
